// DynamicBlock_27857157882171
// MI455X (gfx1250) — matrix operand order repaired and checked by reading, not compiled, not run on hardware
//
#include <hip/hip_runtime.h>
#include <hip/hip_bf16.h>

typedef __attribute__((ext_vector_type(16))) _Float16 v16h;
typedef __attribute__((ext_vector_type(8)))  _Float16 v8h;
typedef __attribute__((ext_vector_type(8)))  float    v8f;

#define SDIM 48
#define CIN 32
#define COUT 32
#define SP3 110592           // 48^3
#define TOT_ELEM 28311552    // B*COUT*S^3
#define KERN_PER_B 27648     // COUT*CIN*27
#define BN_N 884736.0f       // B*S^3

// LDS layout (bytes): kern [27][32][32]f16 @0 (55296), xtile [3][10][50][32]f16 @55296 (96000), sums @151296 (256)
#define LDS_KERN_B   0
#define LDS_X_B      55296
#define LDS_SUM_B    151296
#define LDS_TOTAL_B  151552

// ---- CDNA5 async global->LDS helpers (ASYNCcnt-tracked DMA, bypasses VGPRs) ----
__device__ __forceinline__ void async_load_b128(unsigned lds_byte_off, const void* gaddr) {
    asm volatile("global_load_async_to_lds_b128 %0, %1, off"
                 :: "v"(lds_byte_off), "v"((unsigned long long)(size_t)gaddr)
                 : "memory");
}
__device__ __forceinline__ void wait_async0() {
    asm volatile("s_wait_asynccnt 0" ::: "memory");
}

// ---------------- stage 1: x f32 NCDHW -> f16 NDHWC ----------------
__global__ __launch_bounds__(256) void cvt_x_kernel(const float* __restrict__ x,
                                                    _Float16* __restrict__ xh) {
    int i = blockIdx.x * 256 + threadIdx.x;      // exactly TOT_ELEM threads
    int ci = i & 31;
    int sp = i >> 5;
    int b = sp / SP3;
    int spat = sp - b * SP3;
    xh[i] = (_Float16)x[((size_t)(b * CIN + ci)) * SP3 + spat];
}

// ---------------- stage 2: hypernetwork filter generation ----------------
__global__ __launch_bounds__(256) void gen_kern_kernel(const float* __restrict__ xaux,
                                                       const float* __restrict__ wb,
                                                       const float* __restrict__ wg,
                                                       _Float16* __restrict__ kh) {
    __shared__ float hsh[8];
    int b = blockIdx.x / 108;
    int t = threadIdx.x;
    if (t < 8) {
        float s = 0.f;
        #pragma unroll
        for (int j = 0; j < 31; ++j) s += xaux[b * 31 + j] * wb[t * 31 + j];
        hsh[t] = fmaxf(s, 0.f);
    }
    __syncthreads();
    int g = blockIdx.x * 256 + t;
    int loc = g - b * KERN_PER_B;
    int ci  = loc & 31;
    int co  = (loc >> 5) & 31;
    int tap = loc >> 10;
    int row = (co * CIN + ci) * 27 + tap;
    float v = 0.f;
    #pragma unroll
    for (int j = 0; j < 8; ++j) v += hsh[j] * wg[row * 8 + j];
    kh[g] = (_Float16)v;
}

// ---------------- stage 3: WMMA implicit-GEMM conv ----------------
// block = (b, d, 8 h-rows); 12 waves; each wave: 2 N-tiles x 2 M-tiles x 27 taps
__global__ __launch_bounds__(384) void conv_wmma_kernel(const _Float16* __restrict__ xh,
                                                        const _Float16* __restrict__ kh,
                                                        float* __restrict__ y,
                                                        float* __restrict__ gsum) {
    extern __shared__ char smem[];
    _Float16* lk = (_Float16*)(smem + LDS_KERN_B);
    _Float16* lx = (_Float16*)(smem + LDS_X_B);
    float*  lsum = (float*)(smem + LDS_SUM_B);

    const unsigned lds_base = (unsigned)(size_t)(void*)smem;   // low 32 bits = LDS byte addr

    const int t = threadIdx.x;
    const int bid = blockIdx.x;
    const int b = bid / 288;
    const int rem = bid - b * 288;
    const int d = rem / 6;
    const int h0 = (rem - d * 6) * 8;

    if (t < 64) lsum[t] = 0.f;

    // ---- async DMA: per-sample kernel [tap][co][ci] (all in range) ----
    {
        const char* src = (const char*)(kh + (size_t)b * KERN_PER_B);
        for (int i = t; i < 3456; i += 384)
            async_load_b128(lds_base + LDS_KERN_B + i * 16, src + i * 16);
    }
    // ---- async DMA: x halo tile [dd3][hh10][ww50][ci32], zero-pad OOB ----
    {
        for (int i = t; i < 6000; i += 384) {
            int pos = i >> 2, q = i & 3;
            int dd = pos / 500;  int r2 = pos - dd * 500;
            int hh = r2 / 50;    int ww = r2 - hh * 50;
            int gd = d - 1 + dd, gh = h0 - 1 + hh, gw = ww - 1;
            unsigned dst = lds_base + LDS_X_B + pos * 64 + q * 16;
            if (gd >= 0 && gd < SDIM && gh >= 0 && gh < SDIM && gw >= 0 && gw < SDIM) {
                const _Float16* src = xh + ((((size_t)b * SDIM + gd) * SDIM + gh) * SDIM + gw) * CIN + q * 8;
                async_load_b128(dst, src);
            } else {
                *(uint4*)(smem + LDS_X_B + pos * 64 + q * 16) = make_uint4(0u, 0u, 0u, 0u);
            }
        }
    }
    wait_async0();
    __syncthreads();

    const int wv = t >> 5, lane = t & 31;
    const int hh0 = (wv / 3) * 2;           // first of two output h rows
    const int wt  = wv - (wv / 3) * 3;      // 0..2 (w tile)
    const int n = lane & 15;
    const int kx = lane >> 4;
    const int plo = kx ? 16 : 0;            // contiguous k chunk start

    v8f acc00 = {}, acc01 = {};             // M-tile 0, N-tiles 0/1
    v8f acc10 = {}, acc11 = {};             // M-tile 1, N-tiles 0/1

    #pragma unroll
    for (int tap = 0; tap < 27; ++tap) {
        const int kd = tap / 9;
        const int khh = (tap / 3) % 3;
        const int kw = tap % 3;

        // B fragments for the two h rows
        const _Float16* bp0 = lx + (((kd * 10 + hh0 + khh) * 50 + (wt * 16 + n + kw)) * CIN + kx * 16);
        const _Float16* bp1 = lx + (((kd * 10 + hh0 + 1 + khh) * 50 + (wt * 16 + n + kw)) * CIN + kx * 16);
        v16h bf0 = *(const v16h*)bp0;
        v16h bf1 = *(const v16h*)bp1;

        // A fragments for cout tiles 0 and 1 (shared by both N-tiles)
        const _Float16* a0 = lk + (size_t)(tap * 32 + n) * CIN;
        v8h a0lo = *(const v8h*)(a0 + plo);
        v8h a0hi = *(const v8h*)(a0 + plo + 8);
        v16h af0 = __builtin_shufflevector(a0lo, a0hi, 0,1,2,3,4,5,6,7,8,9,10,11,12,13,14,15);

        const _Float16* a1 = lk + (size_t)(tap * 32 + 16 + n) * CIN;
        v8h a1lo = *(const v8h*)(a1 + plo);
        v8h a1hi = *(const v8h*)(a1 + plo + 8);
        v16h af1 = __builtin_shufflevector(a1lo, a1hi, 0,1,2,3,4,5,6,7,8,9,10,11,12,13,14,15);

        acc00 = __builtin_amdgcn_wmma_f32_16x16x32_f16(false, af0, false, bf0, (short)0, acc00, false, false);
        acc01 = __builtin_amdgcn_wmma_f32_16x16x32_f16(false, af0, false, bf1, (short)0, acc01, false, false);
        acc10 = __builtin_amdgcn_wmma_f32_16x16x32_f16(false, af1, false, bf0, (short)0, acc10, false, false);
        acc11 = __builtin_amdgcn_wmma_f32_16x16x32_f16(false, af1, false, bf1, (short)0, acc11, false, false);
    }

    // store conv output; C/D layout: VGPR r -> M = r + 8*(lane/16), N = lane%16
    {
        size_t base0 = (size_t)b * COUT * SP3 + (size_t)((d * SDIM) + h0 + hh0) * SDIM + wt * 16 + n;
        size_t base1 = base0 + SDIM;   // next h row
        #pragma unroll
        for (int r = 0; r < 8; ++r) {
            int c0 = r + 8 * kx;
            y[base0 + (size_t)c0 * SP3]        = acc00[r];
            y[base1 + (size_t)c0 * SP3]        = acc01[r];
            y[base0 + (size_t)(c0 + 16) * SP3] = acc10[r];
            y[base1 + (size_t)(c0 + 16) * SP3] = acc11[r];
        }
    }

    // BN partial sums
    #pragma unroll
    for (int r = 0; r < 8; ++r) {
        float v0 = acc00[r] + acc01[r];
        float v1 = acc10[r] + acc11[r];
        float q0 = acc00[r] * acc00[r] + acc01[r] * acc01[r];
        float q1 = acc10[r] * acc10[r] + acc11[r] * acc11[r];
        #pragma unroll
        for (int m = 8; m >= 1; m >>= 1) {
            v0 += __shfl_xor(v0, m); q0 += __shfl_xor(q0, m);
            v1 += __shfl_xor(v1, m); q1 += __shfl_xor(q1, m);
        }
        if (n == 0) {
            int c0 = r + 8 * kx;
            atomicAdd(&lsum[c0], v0);       atomicAdd(&lsum[32 + c0], q0);
            atomicAdd(&lsum[c0 + 16], v1);  atomicAdd(&lsum[32 + c0 + 16], q1);
        }
    }
    __syncthreads();
    if (t < 32) {
        atomicAdd(&gsum[t], lsum[t]);
        atomicAdd(&gsum[32 + t], lsum[32 + t]);
    }
}

// ---------------- stage 4: fold batch stats into scale/bias ----------------
__global__ void bn_param_kernel(const float* __restrict__ gsum,
                                const float* __restrict__ gamma,
                                const float* __restrict__ beta,
                                float* __restrict__ sb) {
    int c = threadIdx.x;
    if (c >= COUT) return;
    const float Ninv = 1.0f / BN_N;
    float mean = gsum[c] * Ninv;
    float var  = gsum[32 + c] * Ninv - mean * mean;
    float s = gamma[c] * rsqrtf(var + 1e-5f);
    sb[c] = s;
    sb[32 + c] = beta[c] - mean * s;
}

// ---------------- stage 5: BN + ReLU in place ----------------
__global__ __launch_bounds__(256) void bn_apply_kernel(float* __restrict__ y,
                                                       const float* __restrict__ sb) {
    int i = blockIdx.x * 256 + threadIdx.x;
    int c = (i / SP3) & 31;
    float v = y[i] * sb[c] + sb[32 + c];
    y[i] = fmaxf(v, 0.f);
}

extern "C" void kernel_launch(void* const* d_in, const int* in_sizes, int n_in,
                              void* d_out, int out_size, void* d_ws, size_t ws_size,
                              hipStream_t stream) {
    const float* x     = (const float*)d_in[0];
    const float* xaux  = (const float*)d_in[1];
    const float* wbase = (const float*)d_in[2];
    const float* wgen  = (const float*)d_in[3];
    const float* gamma = (const float*)d_in[4];
    const float* beta  = (const float*)d_in[5];
    float* y = (float*)d_out;

    char* ws = (char*)d_ws;
    float*    gsum = (float*)ws;                               // 64 f32 (sum | sumsq)
    _Float16* kh   = (_Float16*)(ws + 256);                    // 442368 B
    _Float16* xh   = (_Float16*)(ws + 256 + 442368);           // 56.6 MB
    float*    sb   = (float*)(ws + 256 + 442368 + 56623104);   // 64 f32 (scale | bias)

    hipMemsetAsync(gsum, 0, 64 * sizeof(float), stream);

    cvt_x_kernel<<<TOT_ELEM / 256, 256, 0, stream>>>(x, xh);
    gen_kern_kernel<<<8 * KERN_PER_B / 256, 256, 0, stream>>>(xaux, wbase, wgen, kh);

    // 8 samples * 48 d-slices * 6 h-groups = 2304 blocks, 12 waves each
    conv_wmma_kernel<<<2304, 384, LDS_TOTAL_B, stream>>>(xh, kh, y, gsum);

    bn_param_kernel<<<1, 32, 0, stream>>>(gsum, gamma, beta, sb);
    bn_apply_kernel<<<TOT_ELEM / 256, 256, 0, stream>>>(y, sb);
}
